// CrossAttention_27058293965182
// MI455X (gfx1250) — matrix operand order repaired and checked by reading, not compiled, not run on hardware
//
#include <hip/hip_runtime.h>
#include <hip/hip_bf16.h>
#include <math.h>

// ---------------- CDNA5 WMMA fragment types ----------------
typedef __attribute__((ext_vector_type(16))) _Float16 v16h;
typedef __attribute__((ext_vector_type(8)))  _Float16 v8h;
typedef __attribute__((ext_vector_type(8)))  float    v8f;

union FragU {
  v16h v;
  v8h  h[2];
  unsigned u[8];
};

// D = A(16x32) * B(32x16) + C, f16 operands, f32 accumulate.
__device__ __forceinline__ v8f wmma16(v16h a, v16h b, v8f c) {
  return __builtin_amdgcn_wmma_f32_16x16x32_f16(
      false, a, false, b, (short)0, c, false, false);
}

// A-matrix 16x32 fragment from a row-major matrix.
// Lane holds row M = lane&15.  Contiguous k order (same as load_frag_b):
//   elements 0..15 <-> K = k0 + klo + 0..15,  klo = (lane>=16) ? 16 : 0
// rowptr must already point at this lane's row.
__device__ __forceinline__ v16h load_frag_a(const _Float16* rowptr, int k0, int lane) {
  const int klo = (lane & 16) ? 16 : 0;
  FragU f;
  f.h[0] = *(const v8h*)(rowptr + k0 + klo);
  f.h[1] = *(const v8h*)(rowptr + k0 + klo + 8);
  return f.v;
}

// B-matrix 32x16 fragment from B^T stored row-major (column of B contiguous).
// Lane holds column N = lane&15; elements i <-> K = k0 + klo16 + i,
// klo16 = (lane>=16) ? 16 : 0.  colptr points at this lane's B^T row.
__device__ __forceinline__ v16h load_frag_b(const _Float16* colptr, int k0, int lane) {
  const int klo = (lane & 16) ? 16 : 0;
  FragU f;
  f.h[0] = *(const v8h*)(colptr + k0 + klo);
  f.h[1] = *(const v8h*)(colptr + k0 + klo + 8);
  return f.v;
}

__device__ __forceinline__ unsigned pack2h(float a, float b) {
  union { _Float16 h[2]; unsigned u; } t;
  t.h[0] = (_Float16)a;
  t.h[1] = (_Float16)b;
  return t.u;
}

// ---------------- conversion / packing ----------------
__global__ void cvt_f16(const float* __restrict__ in, _Float16* __restrict__ out, int n) {
  int i = blockIdx.x * blockDim.x + threadIdx.x;
  if (i < n) out[i] = (_Float16)in[i];
}

// in[h][d][e]  ->  out[(h*E + e)][d]   (f32 -> f16, per-head transposed pack)
__global__ void pack_wT(const float* __restrict__ in, _Float16* __restrict__ out,
                        int H, int D, int E) {
  int i = blockIdx.x * blockDim.x + threadIdx.x;
  if (i >= H * D * E) return;
  int e = i % E;
  int d = (i / E) % D;
  int h = i / (E * D);
  out[(h * E + e) * D + d] = (_Float16)in[i];
}

// ---------------- f16 GEMM: C[n, col] = sum_d A[n,d] * BT[col,d] ----------------
// 32x64 output tile per wave: 2 A-frags + 4 B-frags -> 8 WMMA per k-step.
// A: [Mrows x 512] row-major f16; BT: [512 x 512] row-major f16 (B transposed).
// trans==0: C row-major [Mrows x 512]; trans==1: C[col*ldcT + n].
__global__ void gemm16_f16(const _Float16* __restrict__ A,
                           const _Float16* __restrict__ BT,
                           _Float16* __restrict__ C,
                           int Mrows, int trans, int ldcT) {
  const int lane = threadIdx.x;
  const int n0 = (blockIdx.y * blockDim.y + threadIdx.y) * 32;
  const int c0 = blockIdx.x * 64;
  if (n0 >= Mrows) return;

  const _Float16* arow0 = A + (size_t)(n0 + (lane & 15)) * 512;
  const _Float16* arow1 = arow0 + (size_t)16 * 512;
  const _Float16* bc = BT + (size_t)(c0 + (lane & 15)) * 512;

  v8f acc[2][4] = {};
  for (int k0 = 0; k0 < 512; k0 += 32) {
    v16h a0 = load_frag_a(arow0, k0, lane);
    v16h a1 = load_frag_a(arow1, k0, lane);
#pragma unroll
    for (int j = 0; j < 4; ++j) {
      v16h b = load_frag_b(bc + (size_t)j * 16 * 512, k0, lane);
      acc[0][j] = wmma16(a0, b, acc[0][j]);
      acc[1][j] = wmma16(a1, b, acc[1][j]);
    }
  }

  const int moff = (lane & 16) ? 8 : 0;
  if (!trans) {
#pragma unroll
    for (int i = 0; i < 2; ++i)
#pragma unroll
      for (int j = 0; j < 4; ++j) {
        _Float16* cp = C + (size_t)(n0 + i * 16 + moff) * 512 + c0 + j * 16 + (lane & 15);
#pragma unroll
        for (int r = 0; r < 8; ++r) cp[(size_t)r * 512] = (_Float16)acc[i][j][r];
      }
  } else {
#pragma unroll
    for (int i = 0; i < 2; ++i)
#pragma unroll
      for (int j = 0; j < 4; ++j) {
        _Float16* cp = C + (size_t)(c0 + j * 16 + (lane & 15)) * ldcT + n0 + i * 16 + moff;
        v8h w;
#pragma unroll
        for (int r = 0; r < 8; ++r) w[r] = (_Float16)acc[i][j][r];
        *(v8h*)cp = w;
      }
  }
}

// ---------------- flash cross-attention ----------------
// S^T = K * Q^T (softmax key-reduction per-lane + one shfl_xor(16)),
// O^T = Vt * P^T (Vt rows are direct A-fragments).  32 queries per wave:
// every K/Vt fragment feeds two query tiles.
__device__ __forceinline__ v16h softmax_update(const v8f& s0, const v8f& s1,
                                               float& m_i, float& l_i,
                                               v8f o[4], bool lohalf) {
  float mloc = s0[0];
#pragma unroll
  for (int r = 1; r < 8; ++r) mloc = fmaxf(mloc, s0[r]);
#pragma unroll
  for (int r = 0; r < 8; ++r) mloc = fmaxf(mloc, s1[r]);
  mloc = fmaxf(mloc, __shfl_xor(mloc, 16, 32));

  const float mnew = fmaxf(m_i, mloc);
  const float corr = __expf(m_i - mnew);
  m_i = mnew;

  float e0[8], e1[8], ssum = 0.0f;
#pragma unroll
  for (int r = 0; r < 8; ++r) { e0[r] = __expf(s0[r] - mnew); ssum += e0[r]; }
#pragma unroll
  for (int r = 0; r < 8; ++r) { e1[r] = __expf(s1[r] - mnew); ssum += e1[r]; }
  ssum += __shfl_xor(ssum, 16, 32);
  l_i = l_i * corr + ssum;

#pragma unroll
  for (int eo = 0; eo < 4; ++eo)
#pragma unroll
    for (int r = 0; r < 8; ++r) o[eo][r] *= corr;

  // Build B-fragment of P^T (32 keys x 16 q): own half + partner half.
  unsigned t0[4], t1[4];
#pragma unroll
  for (int j = 0; j < 4; ++j) {
    t0[j] = pack2h(e0[2 * j], e0[2 * j + 1]);
    t1[j] = pack2h(e1[2 * j], e1[2 * j + 1]);
  }
  FragU pb;
#pragma unroll
  for (int j = 0; j < 4; ++j) {
    unsigned send = lohalf ? t1[j] : t0[j];
    unsigned recv = (unsigned)__shfl_xor((int)send, 16, 32);
    pb.u[j]     = lohalf ? t0[j] : recv;  // keys klo16 + 0..7
    pb.u[4 + j] = lohalf ? recv : t1[j];  // keys klo16 + 8..15
  }
  return pb.v;
}

// Q: [4096 x 512] f16 row-major (cols = h*64+e);  K: [1024 x 512] f16 row-major
// Vt: [512 x 1024] f16 row-major (row = h*64+e, col = key)
// Y: [4096 x 512] f16 row-major output (concat heads)
__global__ void attn16(const _Float16* __restrict__ Q,
                       const _Float16* __restrict__ K,
                       const _Float16* __restrict__ Vt,
                       _Float16* __restrict__ Y) {
  const int lane = threadIdx.x;
  const int q0 = (blockIdx.x * blockDim.y + threadIdx.y) * 32;
  const int h = blockIdx.y;
  const bool lohalf = (lane & 16) == 0;

  // Q columns as B-fragments over e, two query tiles (reused for all keys)
  const _Float16* qcA = Q + (size_t)(q0 + (lane & 15)) * 512 + h * 64;
  const _Float16* qcB = qcA + (size_t)16 * 512;
  const v16h qA0 = load_frag_b(qcA, 0, lane), qA1 = load_frag_b(qcA, 32, lane);
  const v16h qB0 = load_frag_b(qcB, 0, lane), qB1 = load_frag_b(qcB, 32, lane);

  const _Float16* krow = K + (size_t)(lane & 15) * 512 + h * 64;     // + key*512
  const _Float16* vrow = Vt + (size_t)(h * 64 + (lane & 15)) * 1024; // + eo*16*1024 + key

  float mA = -3.0e38f, lA = 0.0f, mB = -3.0e38f, lB = 0.0f;
  v8f oA[4] = {}, oB[4] = {};

  for (int kb = 0; kb < 1024; kb += 32) {
    // ---- scores for 32 keys x 32 queries: K frags shared by both q-tiles ----
    v8f sA0 = {}, sA1 = {}, sB0 = {}, sB1 = {};
    {
      const _Float16* kr0 = krow + (size_t)kb * 512;
      v16h k00 = load_frag_a(kr0, 0, lane);
      v16h k01 = load_frag_a(kr0, 32, lane);
      sA0 = wmma16(k00, qA0, sA0); sA0 = wmma16(k01, qA1, sA0);
      sB0 = wmma16(k00, qB0, sB0); sB0 = wmma16(k01, qB1, sB0);
      const _Float16* kr1 = kr0 + (size_t)16 * 512;
      v16h k10 = load_frag_a(kr1, 0, lane);
      v16h k11 = load_frag_a(kr1, 32, lane);
      sA1 = wmma16(k10, qA0, sA1); sA1 = wmma16(k11, qA1, sA1);
      sB1 = wmma16(k10, qB0, sB1); sB1 = wmma16(k11, qB1, sB1);
    }

    v16h pbA = softmax_update(sA0, sA1, mA, lA, oA, lohalf);
    v16h pbB = softmax_update(sB0, sB1, mB, lB, oB, lohalf);

    // ---- O^T += Vt_tile(16e x 32key) * P^T ; Vt frags shared by both tiles ----
#pragma unroll
    for (int eo = 0; eo < 4; ++eo) {
      v16h va = load_frag_a(vrow + (size_t)eo * 16 * 1024, kb, lane);
      oA[eo] = wmma16(va, pbA, oA[eo]);
      oB[eo] = wmma16(va, pbB, oB[eo]);
    }
  }

  // ---- normalize and store Y row-major (concat-head layout) ----
  const int moff = (lane & 16) ? 8 : 0;
  _Float16* yp = Y + (size_t)(q0 + (lane & 15)) * 512 + h * 64 + moff;
  {
    const float inv = 1.0f / lA;
#pragma unroll
    for (int eo = 0; eo < 4; ++eo) {
      v8h w;
#pragma unroll
      for (int r = 0; r < 8; ++r) w[r] = (_Float16)(oA[eo][r] * inv);
      *(v8h*)(yp + eo * 16) = w;
    }
  }
  {
    const float inv = 1.0f / lB;
    _Float16* yq = yp + (size_t)16 * 512;
#pragma unroll
    for (int eo = 0; eo < 4; ++eo) {
      v8h w;
#pragma unroll
      for (int r = 0; r < 8; ++r) w[r] = (_Float16)(oB[eo][r] * inv);
      *(v8h*)(yq + eo * 16) = w;
    }
  }
}

// ---------------- output projection + positional encoding ----------------
// Out[n][j] = sum_k Y[n][k] * Wout[k][j] + PE(n, j); 32x64 tile per wave.
__global__ void outgemm_pe(const _Float16* __restrict__ Y,
                           const _Float16* __restrict__ WoutT,
                           float* __restrict__ Out) {
  const int lane = threadIdx.x;
  const int n0 = (blockIdx.y * blockDim.y + threadIdx.y) * 32;
  const int j0 = blockIdx.x * 64;

  const _Float16* ar0 = Y + (size_t)(n0 + (lane & 15)) * 512;
  const _Float16* ar1 = ar0 + (size_t)16 * 512;
  const _Float16* bc = WoutT + (size_t)(j0 + (lane & 15)) * 512;

  v8f acc[2][4] = {};
  for (int k0 = 0; k0 < 512; k0 += 32) {
    v16h a0 = load_frag_a(ar0, k0, lane);
    v16h a1 = load_frag_a(ar1, k0, lane);
#pragma unroll
    for (int j = 0; j < 4; ++j) {
      v16h b = load_frag_b(bc + (size_t)j * 16 * 512, k0, lane);
      acc[0][j] = wmma16(a0, b, acc[0][j]);
      acc[1][j] = wmma16(a1, b, acc[1][j]);
    }
  }

  const int moff = (lane & 16) ? 8 : 0;
#pragma unroll
  for (int jt = 0; jt < 4; ++jt) {
    const int j = j0 + jt * 16 + (lane & 15);
    // freq[i] = exp(i * ln(10000)/512); x = n / freq
    const float invfreq = __expf(-(float)(j >> 1) * 0.017988946039015984f);
    const bool is_cos = (j & 1) != 0;
#pragma unroll
    for (int i = 0; i < 2; ++i) {
      float* op = Out + (size_t)(n0 + i * 16 + moff) * 512 + j;
#pragma unroll
      for (int r = 0; r < 8; ++r) {
        float x = (float)(n0 + i * 16 + moff + r) * invfreq;
        float pe = is_cos ? cosf(x) : sinf(x);
        op[(size_t)r * 512] = acc[i][jt][r] + pe;
      }
    }
  }
}

// ---------------- host launch ----------------
extern "C" void kernel_launch(void* const* d_in, const int* in_sizes, int n_in,
                              void* d_out, int out_size, void* d_ws, size_t ws_size,
                              hipStream_t stream) {
  (void)in_sizes; (void)n_in; (void)out_size; (void)ws_size;
  const int N = 4096, M = 1024, H = 8, D = 512, DH = 64;

  const float* video = (const float*)d_in[0];  // [N, 512]
  const float* text  = (const float*)d_in[1];  // [M, 512]
  const float* Wq    = (const float*)d_in[2];  // [H, 512, 64]
  const float* Wk    = (const float*)d_in[3];
  const float* Wv    = (const float*)d_in[4];
  const float* Wout  = (const float*)d_in[5];  // [512, 512]
  float* out = (float*)d_out;                  // [N, 512]

  _Float16* p = (_Float16*)d_ws;
  _Float16* videoH = p; p += (size_t)N * D;   // 4096x512
  _Float16* textH  = p; p += (size_t)M * D;   // 1024x512
  _Float16* WqT    = p; p += (size_t)D * D;   // 512x512 (col=h*64+e, row-major over d)
  _Float16* WkT    = p; p += (size_t)D * D;
  _Float16* WvT    = p; p += (size_t)D * D;
  _Float16* WoT    = p; p += (size_t)D * D;   // WoutT[j][k]
  _Float16* Qh     = p; p += (size_t)N * D;   // 4096x512 row-major
  _Float16* Kh     = p; p += (size_t)M * D;   // 1024x512 row-major
  _Float16* VtH    = p; p += (size_t)D * M;   // 512x1024 (row = h*64+e)
  _Float16* Yh     = p; p += (size_t)N * D;   // 4096x512 row-major

  // 1) convert activations, pack weights transposed
  cvt_f16<<<(N * D + 255) / 256, 256, 0, stream>>>(video, videoH, N * D);
  cvt_f16<<<(M * D + 255) / 256, 256, 0, stream>>>(text, textH, M * D);
  pack_wT<<<(H * D * DH + 255) / 256, 256, 0, stream>>>(Wq, WqT, H, D, DH);
  pack_wT<<<(H * D * DH + 255) / 256, 256, 0, stream>>>(Wk, WkT, H, D, DH);
  pack_wT<<<(H * D * DH + 255) / 256, 256, 0, stream>>>(Wv, WvT, H, D, DH);
  pack_wT<<<(D * D + 255) / 256, 256, 0, stream>>>(Wout, WoT, 1, D, D);

  // 2) QKV projections (f16 WMMA GEMMs, 32x64 tiles); V written transposed
  dim3 blk(32, 4);
  gemm16_f16<<<dim3(D / 64, N / 128), blk, 0, stream>>>(videoH, WqT, Qh, N, 0, 0);
  gemm16_f16<<<dim3(D / 64, M / 128), blk, 0, stream>>>(textH, WkT, Kh, M, 0, 0);
  gemm16_f16<<<dim3(D / 64, M / 128), blk, 0, stream>>>(textH, WvT, VtH, M, 1, M);

  // 3) flash cross-attention (per head, 32-query tiles per wave)
  attn16<<<dim3(N / 128, H), blk, 0, stream>>>(Qh, Kh, VtH, Yh);

  // 4) output projection + fused positional encoding (f32 out)
  outgemm_pe<<<dim3(D / 64, N / 128), blk, 0, stream>>>(Yh, WoT, out);
}
